// GATv2Model_84207128805735
// MI455X (gfx1250) — matrix operand order repaired and checked by reading, not compiled, not run on hardware
//
#include <hip/hip_runtime.h>

// ---------------- problem constants ----------------
#define NN   50000LL     // nodes
#define EE   500000LL    // raw edges
#define ETOT 550000LL    // edges + self loops
#define FIN  128
#define FE   16
#define HH   8
#define HC1  256
#define HC2  128
#define NSLOPE 0.2f
#define BN_EPS 1e-5f

typedef __bf16 bf16_t;
typedef bf16_t bf16x8  __attribute__((ext_vector_type(8)));
typedef bf16_t bf16x16 __attribute__((ext_vector_type(16)));
typedef float  f32x8   __attribute__((ext_vector_type(8)));

// ---------------- helpers ----------------
__device__ __forceinline__ void edge_sd(const int* ei, long long e, int& s, int& d) {
    if (e < EE) { s = ei[e]; d = ei[EE + e]; }
    else        { s = d = (int)(e - EE); }
}

__device__ __forceinline__ void atomicMaxF(float* addr, float value) {
    // bit-trick float max: works across sign combinations with -inf init
    if (value >= 0.0f) atomicMax((int*)addr, __float_as_int(value));
    else               atomicMin((unsigned int*)addr, __float_as_uint(value));
}

// ---------------- utility kernels ----------------
__global__ void k_fill(float* p, long long n, float v) {
    long long t = (long long)blockIdx.x * blockDim.x + threadIdx.x;
    if (t < n) p[t] = v;
}

__global__ void k_cvt_bf16(const float* in, bf16_t* out, long long n) {
    long long t = (long long)blockIdx.x * blockDim.x + threadIdx.x;
    if (t < n) out[t] = (bf16_t)in[t];
}

// mean of edge_attr over rows (partial sums -> atomicAdd)
__global__ void k_ea_accum(const float* ea, float* acc) {
    int c   = threadIdx.x & 15;
    int grp = threadIdx.x >> 4;                       // 16 row-groups per block
    long long row = (long long)blockIdx.x * 16 + grp;
    long long stride = (long long)gridDim.x * 16;
    float s = 0.f;
    for (long long r = row; r < EE; r += stride) s += ea[r * FE + c];
    atomicAdd(&acc[c], s);
}

__global__ void k_ea_final(const float* acc, float* mean) {
    int c = threadIdx.x;
    if (c < FE) mean[c] = acc[c] * (1.0f / (float)EE);
}

// pack W[K,Nn] (f32) into bf16 WMMA B-fragment order:
// Bp[((nt*KT+kt)*32 + lane)*16 + j] = W[(kt*32 + (lane>>4)*8 + (j&7) + (j>>3)*16)*Nn + nt*16 + (lane&15)]
__global__ void k_pack_b(const float* W, bf16_t* Bp, int K, int Nn) {
    long long tid = (long long)blockIdx.x * blockDim.x + threadIdx.x;
    long long total = (long long)K * Nn;
    if (tid >= total) return;
    int j    = (int)(tid & 15);
    int lane = (int)((tid >> 4) & 31);
    long long t2 = tid >> 9;                          // nt*KT + kt
    int KT = K >> 5;
    int kt = (int)(t2 % KT);
    int nt = (int)(t2 / KT);
    int col = (nt << 4) + (lane & 15);
    int k   = (kt << 5) + ((lane >> 4) << 3) + (j & 7) + ((j >> 3) << 4);
    Bp[tid] = (bf16_t)W[(long long)k * Nn + col];
}

// ---------------- WMMA GEMM, 5(M)x2(N) register-blocked ----------------
// C[M,Nn] = A[M,K](bf16, row major) * W(bf16 packed) + bias
// Per wave: 80 rows x 32 cols; 10 WMMAs per K-step on 5 A-frags + 2 B-frags
// (~23 flops/byte). Live state 136 VGPRs -> no spills, room for pipelined loads.
// Requires (M/16)%5==0 and (Nn/16)%2==0.
__global__ void __launch_bounds__(256, 1)
k_gemm_bf16(const bf16_t* __restrict__ A, const bf16_t* __restrict__ Bp,
            const float* __restrict__ bias, float* __restrict__ C,
            int M, int K, int Nn) {
    int lane = threadIdx.x & 31;
    long long wave = (((long long)blockIdx.x * blockDim.x + threadIdx.x) >> 5);
    int NT  = Nn >> 4;
    int KT  = K >> 5;
    int NT2 = NT >> 1;                                 // N groups of 2 tiles
    long long nwaves = (long long)((M >> 4) / 5) * NT2;
    if (wave >= nwaves) return;                        // wave-uniform exit: EXEC all-1 inside
    int mt5 = (int)(wave / NT2);
    int nt2 = (int)(wave % NT2);
    int hi  = lane >> 4;
    int r0  = mt5 * 80 + (lane & 15);                  // base row for sub-tile i=0

    const bf16_t* arow  = A + (long long)r0 * K + hi * 8;
    const bf16_t* bbase = Bp + ((long long)(nt2 * 2) * KT * 32 + lane) * 16;
    const long long aStride = (long long)16 * K;       // between M sub-tiles
    const long long bStride = (long long)KT * 32 * 16; // between N sub-tiles

    f32x8 acc[5][2];
#pragma unroll
    for (int i = 0; i < 5; ++i)
#pragma unroll
        for (int j = 0; j < 2; ++j) acc[i][j] = (f32x8){};

    for (int kt = 0; kt < KT; ++kt) {
        bf16x16 a[5], b[2];
#pragma unroll
        for (int i = 0; i < 5; ++i) {
            const bf16_t* ap = arow + i * aStride;
            bf16x8 alo = *(const bf16x8*)(ap);
            bf16x8 ahi = *(const bf16x8*)(ap + 16);
#pragma unroll
            for (int q = 0; q < 8; ++q) { a[i][q] = alo[q]; a[i][q + 8] = ahi[q]; }
        }
#pragma unroll
        for (int j = 0; j < 2; ++j)
            b[j] = *(const bf16x16*)(bbase + j * bStride);
        arow  += 32;       // next K block in A
        bbase += 32 * 16;  // next K block in packed B
#pragma unroll
        for (int i = 0; i < 5; ++i)
#pragma unroll
            for (int j = 0; j < 2; ++j)
                acc[i][j] = __builtin_amdgcn_wmma_f32_16x16x32_bf16(
                                false, a[i], false, b[j], (short)0, acc[i][j],
                                false, false);
    }

    int colD = lane & 15;
#pragma unroll
    for (int j = 0; j < 2; ++j) {
        int col = (nt2 * 2 + j) * 16 + colD;
        float bv = bias ? bias[col] : 0.f;
#pragma unroll
        for (int i = 0; i < 5; ++i) {
            int mrow0 = mt5 * 80 + i * 16 + (hi << 3);
#pragma unroll
            for (int v = 0; v < 8; ++v)
                C[(long long)(mrow0 + v) * Nn + col] = acc[i][j][v] + bv;
        }
    }
}

// ---------------- attention: pass 1 (logits + segment max) ----------------
template<int C, int H>
__global__ void k_edge_logit(const int* __restrict__ ei, const float* __restrict__ eattr,
                             const float* __restrict__ ea_mean,
                             const float* __restrict__ xs, const float* __restrict__ xd,
                             const float* __restrict__ We, const float* __restrict__ att,
                             float* __restrict__ logit, float* __restrict__ lmax) {
    constexpr int HC = C * H;
    __shared__ float sWe[FE * HC];
    __shared__ float sAtt[HC];
    for (int i = threadIdx.x; i < FE * HC; i += blockDim.x) sWe[i] = We[i];
    for (int i = threadIdx.x; i < HC; i += blockDim.x) sAtt[i] = att[i];
    __syncthreads();

    long long tid = (long long)blockIdx.x * blockDim.x + threadIdx.x;
    if (tid >= ETOT * (long long)H) return;
    long long e = tid / H;
    int h = (int)(tid % H);
    int s, d; edge_sd(ei, e, s, d);

    float ea[FE];
    const float* ap = (e < EE) ? (eattr + e * FE) : ea_mean;
#pragma unroll
    for (int k = 0; k < FE; ++k) ea[k] = ap[k];

    const float* xsp = xs + (long long)s * HC + h * C;
    const float* xdp = xd + (long long)d * HC + h * C;
    float acc = 0.f;
#pragma unroll 4
    for (int c = 0; c < C; ++c) {
        float m = xsp[c] + xdp[c];
#pragma unroll
        for (int k = 0; k < FE; ++k) m += ea[k] * sWe[k * HC + h * C + c];
        m = (m > 0.f) ? m : m * NSLOPE;
        acc += m * sAtt[h * C + c];
    }
    logit[tid] = acc;
    atomicMaxF(&lmax[(long long)d * H + h], acc);
}

// ---------------- attention: pass 2 (exp + segment sum) ----------------
template<int H>
__global__ void k_edge_exp(const int* __restrict__ ei, float* __restrict__ logit_ex,
                           const float* __restrict__ lmax, float* __restrict__ denom) {
    long long tid = (long long)blockIdx.x * blockDim.x + threadIdx.x;
    if (tid >= ETOT * (long long)H) return;
    long long e = tid / H;
    int h = (int)(tid % H);
    int s, d; edge_sd(ei, e, s, d); (void)s;
    float ex = __expf(logit_ex[tid] - lmax[(long long)d * H + h]);
    logit_ex[tid] = ex;
    atomicAdd(&denom[(long long)d * H + h], ex);
}

// ---------------- attention: pass 3 (alpha-weighted scatter) ----------------
template<int C, int H>
__global__ void k_edge_scatter(const int* __restrict__ ei, const float* __restrict__ xs,
                               const float* __restrict__ ex, const float* __restrict__ denom,
                               float* __restrict__ out) {
    constexpr int HC = C * H;
    long long tid = (long long)blockIdx.x * blockDim.x + threadIdx.x;
    if (tid >= ETOT * (long long)HC) return;
    long long e = tid / HC;
    int f = (int)(tid % HC);
    int h = f / C;
    int s, d; edge_sd(ei, e, s, d);
    float alpha = ex[e * H + h] / (denom[(long long)d * H + h] + 1e-16f);
    atomicAdd(&out[(long long)d * HC + f], alpha * xs[(long long)s * HC + f]);
}

// ---------------- batchnorm stats + fused BN/ELU ----------------
__global__ void k_bn_stats(const float* __restrict__ x, float* gsum, float* gss,
                           int HC, int rowsPer) {
    int c = threadIdx.x;                               // blockDim == HC
    long long r0 = (long long)blockIdx.x * rowsPer;
    long long r1 = r0 + rowsPer; if (r1 > NN) r1 = NN;
    float s = 0.f, ss = 0.f;
    for (long long r = r0; r < r1; ++r) {
        float v = x[r * HC + c];
        s += v; ss += v * v;
    }
    atomicAdd(&gsum[c], s);
    atomicAdd(&gss[c], ss);
}

__global__ void k_bn_elu(const float* __restrict__ x, const float* __restrict__ gsum,
                         const float* __restrict__ gss, const float* __restrict__ g,
                         const float* __restrict__ bt, bf16_t* obf, float* of, int HC) {
    long long tid = (long long)blockIdx.x * blockDim.x + threadIdx.x;
    if (tid >= NN * (long long)HC) return;
    int c = (int)(tid % HC);
    float inv_n = 1.0f / (float)NN;
    float mu  = gsum[c] * inv_n;
    float var = gss[c] * inv_n - mu * mu;
    float y = g[c] * (x[tid] - mu) * rsqrtf(var + BN_EPS) + bt[c];
    y = (y > 0.f) ? y : expm1f(y);
    if (obf) obf[tid] = (bf16_t)y;
    if (of)  of[tid]  = y;
}

// ---------------- conv3 GEMV (OUT=1, two weight sets) ----------------
__global__ void k_gemv3(const float* __restrict__ h2, const float* __restrict__ W3l,
                        const float* __restrict__ b3l, const float* __restrict__ W3r,
                        const float* __restrict__ b3r, float* xl3, float* xr3) {
    __shared__ float sWl[HC2], sWr[HC2];
    for (int i = threadIdx.x; i < HC2; i += blockDim.x) { sWl[i] = W3l[i]; sWr[i] = W3r[i]; }
    __syncthreads();
    long long n = (long long)blockIdx.x * blockDim.x + threadIdx.x;
    if (n >= NN) return;
    const float* hp = h2 + n * HC2;
    float al = 0.f, ar = 0.f;
#pragma unroll 4
    for (int k = 0; k < HC2; ++k) { float v = hp[k]; al += v * sWl[k]; ar += v * sWr[k]; }
    xl3[n] = al + b3l[0];
    xr3[n] = ar + b3r[0];
}

__global__ void k_out_final(const float* out3, const float* bo3, float* dout) {
    long long n = (long long)blockIdx.x * blockDim.x + threadIdx.x;
    if (n < NN) dout[n] = out3[n] + bo3[0];
}

// ---------------- launch ----------------
static inline unsigned gblk(long long n) { return (unsigned)((n + 255) / 256); }

extern "C" void kernel_launch(void* const* d_in, const int* in_sizes, int n_in,
                              void* d_out, int out_size, void* d_ws, size_t ws_size,
                              hipStream_t stream) {
    (void)in_sizes; (void)n_in; (void)out_size; (void)ws_size;

    const float* x     = (const float*)d_in[0];
    const int*   ei    = (const int*)  d_in[1];
    const float* eattr = (const float*)d_in[2];
    const float* W1  = (const float*)d_in[3];  const float* b1  = (const float*)d_in[4];
    const float* We1 = (const float*)d_in[5];  const float* att1= (const float*)d_in[6];
    /* bo1 d_in[7] cancels under BN */
    const float* g1  = (const float*)d_in[8];  const float* bt1 = (const float*)d_in[9];
    const float* W2  = (const float*)d_in[10]; const float* b2  = (const float*)d_in[11];
    const float* We2 = (const float*)d_in[12]; const float* att2= (const float*)d_in[13];
    /* bo2 d_in[14] cancels under BN */
    const float* g2  = (const float*)d_in[15]; const float* bt2 = (const float*)d_in[16];
    const float* W3l = (const float*)d_in[17]; const float* b3l = (const float*)d_in[18];
    const float* W3r = (const float*)d_in[19]; const float* b3r = (const float*)d_in[20];
    const float* We3 = (const float*)d_in[21]; const float* att3= (const float*)d_in[22];
    const float* bo3 = (const float*)d_in[23];
    float* dout = (float*)d_out;

    // -------- workspace carve (bump allocator, 256B aligned) --------
    char* p = (char*)d_ws;
    auto take = [&](size_t bytes) -> void* {
        void* r = (void*)p;
        p += (bytes + 255) & ~(size_t)255;
        return r;
    };
    float*  ea_acc  = (float*) take(FE * 4);
    float*  ea_mean = (float*) take(FE * 4);
    float*  bn_sum  = (float*) take(HC1 * 4);
    float*  bn_ss   = (float*) take(HC1 * 4);
    bf16_t* x_bf    = (bf16_t*)take(NN * FIN * 2);
    bf16_t* W1p     = (bf16_t*)take((size_t)FIN * HC1 * 2);
    bf16_t* W2p     = (bf16_t*)take((size_t)HC1 * HC2 * 2);
    float*  xl1     = (float*) take(NN * HC1 * 4);   // reused as h2 f32 later
    float*  outac   = (float*) take(NN * HC1 * 4);   // reused across all layers
    float*  exbuf   = (float*) take(ETOT * HH * 4);  // logits -> exp, layers 1/2
    float*  lmax    = (float*) take(NN * HH * 4);
    float*  denom   = (float*) take(NN * HH * 4);
    bf16_t* h1_bf   = (bf16_t*)take(NN * HC1 * 2);
    float*  xl2     = (float*) take(NN * HC2 * 4);
    float*  xl3     = (float*) take(NN * 4);
    float*  xr3     = (float*) take(NN * 4);
    float*  lmax3   = (float*) take(NN * 4);
    float*  denom3  = (float*) take(NN * 4);
    float*  ex3     = (float*) take(ETOT * 4);
    float*  h2      = xl1;                           // reuse (dead after layer-1 scatter)
    float*  out3    = outac;                         // reuse

    const float NEG_INF = -__builtin_huge_valf();

    // -------- edge_attr mean (self-loop fill) --------
    k_fill<<<1, 32, 0, stream>>>(ea_acc, FE, 0.f);
    k_ea_accum<<<512, 256, 0, stream>>>(eattr, ea_acc);
    k_ea_final<<<1, 32, 0, stream>>>(ea_acc, ea_mean);

    // -------- precision prep: bf16 activations + packed weights --------
    k_cvt_bf16<<<gblk(NN * FIN), 256, 0, stream>>>(x, x_bf, NN * FIN);
    k_pack_b<<<gblk((long long)FIN * HC1), 256, 0, stream>>>(W1, W1p, FIN, HC1);
    k_pack_b<<<gblk((long long)HC1 * HC2), 256, 0, stream>>>(W2, W2p, HC1, HC2);

    // ================= layer 1 =================
    k_fill<<<gblk(NN * HH),  256, 0, stream>>>(lmax,  NN * HH, NEG_INF);
    k_fill<<<gblk(NN * HH),  256, 0, stream>>>(denom, NN * HH, 0.f);
    k_fill<<<gblk(NN * HC1), 256, 0, stream>>>(outac, NN * HC1, 0.f);

    {   // xl1 = x @ W1 + b1 via bf16 WMMA (5x2 register-blocked)
        long long nwaves = ((NN / 16) / 5) * ((HC1 / 16) / 2);   // 625*8 = 5000
        k_gemm_bf16<<<gblk(nwaves * 32), 256, 0, stream>>>(x_bf, W1p, b1, xl1,
                                                           (int)NN, FIN, HC1);
    }
    k_edge_logit<32, HH><<<gblk(ETOT * HH), 256, 0, stream>>>(
        ei, eattr, ea_mean, xl1, xl1, We1, att1, exbuf, lmax);
    k_edge_exp<HH><<<gblk(ETOT * HH), 256, 0, stream>>>(ei, exbuf, lmax, denom);
    k_edge_scatter<32, HH><<<gblk(ETOT * HC1), 256, 0, stream>>>(
        ei, xl1, exbuf, denom, outac);

    k_fill<<<1, 256, 0, stream>>>(bn_sum, HC1, 0.f);
    k_fill<<<1, 256, 0, stream>>>(bn_ss,  HC1, 0.f);
    k_bn_stats<<<256, HC1, 0, stream>>>(outac, bn_sum, bn_ss, HC1, (int)((NN + 255) / 256));
    k_bn_elu<<<gblk(NN * HC1), 256, 0, stream>>>(outac, bn_sum, bn_ss, g1, bt1,
                                                 h1_bf, (float*)nullptr, HC1);

    // ================= layer 2 =================
    k_fill<<<gblk(NN * HH),  256, 0, stream>>>(lmax,  NN * HH, NEG_INF);
    k_fill<<<gblk(NN * HH),  256, 0, stream>>>(denom, NN * HH, 0.f);
    k_fill<<<gblk(NN * HC2), 256, 0, stream>>>(outac, NN * HC2, 0.f);

    {   // xl2 = h1 @ W2 + b2 (5x2 register-blocked)
        long long nwaves = ((NN / 16) / 5) * ((HC2 / 16) / 2);   // 625*4 = 2500
        k_gemm_bf16<<<gblk(nwaves * 32), 256, 0, stream>>>(h1_bf, W2p, b2, xl2,
                                                           (int)NN, HC1, HC2);
    }
    k_edge_logit<16, HH><<<gblk(ETOT * HH), 256, 0, stream>>>(
        ei, eattr, ea_mean, xl2, xl2, We2, att2, exbuf, lmax);
    k_edge_exp<HH><<<gblk(ETOT * HH), 256, 0, stream>>>(ei, exbuf, lmax, denom);
    k_edge_scatter<16, HH><<<gblk(ETOT * HC2), 256, 0, stream>>>(
        ei, xl2, exbuf, denom, outac);

    k_fill<<<1, 256, 0, stream>>>(bn_sum, HC2, 0.f);
    k_fill<<<1, 256, 0, stream>>>(bn_ss,  HC2, 0.f);
    k_bn_stats<<<256, HC2, 0, stream>>>(outac, bn_sum, bn_ss, HC2, (int)((NN + 255) / 256));
    k_bn_elu<<<gblk(NN * HC2), 256, 0, stream>>>(outac, bn_sum, bn_ss, g2, bt2,
                                                 (bf16_t*)nullptr, h2, HC2);

    // ================= layer 3 (1 head, OUT=1) =================
    k_gemv3<<<gblk(NN), 256, 0, stream>>>(h2, W3l, b3l, W3r, b3r, xl3, xr3);

    k_fill<<<gblk(NN), 256, 0, stream>>>(lmax3,  NN, NEG_INF);
    k_fill<<<gblk(NN), 256, 0, stream>>>(denom3, NN, 0.f);
    k_fill<<<gblk(NN), 256, 0, stream>>>(out3,   NN, 0.f);

    k_edge_logit<1, 1><<<gblk(ETOT), 256, 0, stream>>>(
        ei, eattr, ea_mean, xl3, xr3, We3, att3, ex3, lmax3);
    k_edge_exp<1><<<gblk(ETOT), 256, 0, stream>>>(ei, ex3, lmax3, denom3);
    k_edge_scatter<1, 1><<<gblk(ETOT), 256, 0, stream>>>(ei, xl3, ex3, denom3, out3);

    k_out_final<<<gblk(NN), 256, 0, stream>>>(out3, bo3, dout);
}
